// MolVisGNN_45552423141683
// MI455X (gfx1250) — matrix operand order repaired and checked by reading, not compiled, not run on hardware
//
#include <hip/hip_runtime.h>
#include <hip/hip_bf16.h>
#include <math.h>

// ---------------------------------------------------------------------------
// MI455X (gfx1250) implementation of the MolVisGNN forward pass.
// Dense matmuls (gate MLP, SAGE linear layers) -> v_wmma_f32_16x16x32_bf16.
// Everything else (pools, batchnorm stats, tiny conv net, edge scatter) is
// bandwidth-bound and uses plain VALU + f32 L2 atomics.
// ---------------------------------------------------------------------------

typedef __attribute__((ext_vector_type(16))) __bf16 v16bf;
typedef __attribute__((ext_vector_type(8)))  float  v8f;

#define ND 30000
#define NP 20000
#define NE 600000

// ---------------- helpers ----------------

// pack two f32 into packed bf16x2 (truncating): one v_perm_b32.
// dst = { b[31:16], a[31:16] }  ->  bytes {7,6} of S0=b, bytes {3,2} of S1=a.
__device__ __forceinline__ unsigned pack_bf16x2(float a, float b) {
#if __has_builtin(__builtin_amdgcn_cvt_pk_bf16_f32)
  auto r = __builtin_amdgcn_cvt_pk_bf16_f32(a, b);
  union { decltype(r) v; unsigned u; } c;
  c.v = r;
  return c.u;
#else
  union { float f; unsigned u; } x, y;
  x.f = a; y.f = b;
  return __builtin_amdgcn_perm(y.u, x.u, 0x07060302u);
#endif
}

// A fragment: 16x32 bf16, ISA 7.12.2 layout. lane = m(0..15) + 16*half.
// Per lane this is two runs of 8 consecutive floats: k = half*8+{0..7} and
// k = 16+half*8+{0..7}  ->  four float4 (global_load_b128) fetches.
__device__ __forceinline__ v16bf load_a_frag(const float* __restrict__ arow_kb_half8) {
  const float4* pa = reinterpret_cast<const float4*>(arow_kb_half8);
  float4 a0 = pa[0], a1 = pa[1];   // k 0..7   (relative)
  float4 a2 = pa[4], a3 = pa[5];   // k 16..23 (relative)
  union { v16bf v; unsigned u[8]; } fr;
  fr.u[0] = pack_bf16x2(a0.x, a0.y); fr.u[1] = pack_bf16x2(a0.z, a0.w);
  fr.u[2] = pack_bf16x2(a1.x, a1.y); fr.u[3] = pack_bf16x2(a1.z, a1.w);
  fr.u[4] = pack_bf16x2(a2.x, a2.y); fr.u[5] = pack_bf16x2(a2.z, a2.w);
  fr.u[6] = pack_bf16x2(a3.x, a3.y); fr.u[7] = pack_bf16x2(a3.z, a3.w);
  return fr.v;
}

// B fragment: 32x16 bf16 (K x N). lane = n(0..15) + 16*half; per lane two runs
// of 8 consecutive floats: k = half*8+{0..7} and k = 16+half*8+{0..7}.
__device__ __forceinline__ v16bf load_b_frag(const float* __restrict__ wcol_kb_half16) {
  const float4* pb = reinterpret_cast<const float4*>(wcol_kb_half16);
  float4 b0 = pb[0], b1 = pb[1], b2 = pb[4], b3 = pb[5];
  union { v16bf v; unsigned u[8]; } fr;
  fr.u[0] = pack_bf16x2(b0.x, b0.y); fr.u[1] = pack_bf16x2(b0.z, b0.w);
  fr.u[2] = pack_bf16x2(b1.x, b1.y); fr.u[3] = pack_bf16x2(b1.z, b1.w);
  fr.u[4] = pack_bf16x2(b2.x, b2.y); fr.u[5] = pack_bf16x2(b2.z, b2.w);
  fr.u[6] = pack_bf16x2(b3.x, b3.y); fr.u[7] = pack_bf16x2(b3.z, b3.w);
  return fr.v;
}

// ---------------- WMMA GEMM:  out = act( A0@W0^T [+ A1@W1^T] + bias ) -------
// A* : N x K row-major f32 (converted to bf16 in-register)
// W* : M x K row-major f32;  CT = ceil(M/16) column tiles per wave (A reuse)
// act: 0=none 1=relu 2=leaky(0.5)
// Out-of-range rows/cols are CLAMPED on load (branchless): a bad A row only
// pollutes its own D row, a bad B col only its own D col; neither is stored.
template<int CT>
__global__ __launch_bounds__(128)
void wmma_gemm_kernel(const float* __restrict__ A0, const float* __restrict__ W0, int K0,
                      const float* __restrict__ A1, const float* __restrict__ W1, int K1,
                      const float* __restrict__ bias,
                      float* __restrict__ out, int N, int M, int act) {
  const int lane = threadIdx.x;                     // wave32
  const int row_base = (blockIdx.x * 4 + threadIdx.y) * 16;
  if (row_base >= N) return;
  const int n = lane & 15, half = lane >> 4;
  const int row = min(row_base + n, N - 1);         // m == n bits of lane

  v8f acc[CT];
#pragma unroll
  for (int ct = 0; ct < CT; ++ct) acc[ct] = (v8f){};

  const float* wcol[CT];

#pragma unroll
  for (int pass = 0; pass < 2; ++pass) {
    const float* A = pass ? A1 : A0;
    const float* W = pass ? W1 : W0;
    const int    K = pass ? K1 : K0;
    if (!A) continue;                                // uniform branch
    const float* arow = A + (size_t)row * K + half * 8;
#pragma unroll
    for (int ct = 0; ct < CT; ++ct)
      wcol[ct] = W + (size_t)min(ct * 16 + n, M - 1) * K + half * 8;

    for (int kb = 0; kb < K; kb += 32) {
      __builtin_prefetch(arow + kb + 32, 0, 1);
      v16bf af = load_a_frag(arow + kb);
#pragma unroll
      for (int ct = 0; ct < CT; ++ct) {
        v16bf bf = load_b_frag(wcol[ct] + kb);
        acc[ct] = __builtin_amdgcn_wmma_f32_16x16x32_bf16(false, af, false, bf,
                                                          (short)0, acc[ct],
                                                          false, false);
      }
    }
  }

  // C/D layout: VGPR vi -> row m = vi + 8*half, col = ct*16 + (lane&15)
#pragma unroll
  for (int ct = 0; ct < CT; ++ct) {
    int col = ct * 16 + n;
    if (col < M) {
      float bv = bias ? bias[col] : 0.0f;
#pragma unroll
      for (int vi = 0; vi < 8; ++vi) {
        int r = row_base + half * 8 + vi;
        if (r < N) {
          float f = acc[ct][vi] + bv;
          if (act == 1)      f = fmaxf(f, 0.0f);
          else if (act == 2) f = (f > 0.0f) ? f : 0.5f * f;
          out[(size_t)r * M + col] = f;
        }
      }
    }
  }
}

// ---------------- elementwise / reduction kernels ----------------

__global__ __launch_bounds__(256)
void pool4_kernel(const float* __restrict__ xa, const float* __restrict__ xb,
                  float* __restrict__ d1, float* __restrict__ d2) {
  int gid = blockIdx.x * 256 + threadIdx.x;
  if (gid >= ND * 128) return;
  int i = gid >> 7, j = gid & 127;
  const float* p = xa + (size_t)i * 512 + j * 4;
  d1[gid] = 0.25f * (p[0] + p[1] + p[2] + p[3]);
  p = xb + (size_t)i * 512 + j * 4;
  d2[gid] = 0.25f * (p[0] + p[1] + p[2] + p[3]);
}

__global__ __launch_bounds__(256)
void gate_softmax_kernel(const float* __restrict__ h0, const float* __restrict__ h1,
                         const float* __restrict__ h2, float* __restrict__ g) {
  int i = blockIdx.x * 256 + threadIdx.x;
  if (i >= ND) return;
  const float* hs[3] = { h0, h1, h2 };
  float s[3];
#pragma unroll
  for (int b = 0; b < 3; ++b) {
    float m = 0.f;
#pragma unroll
    for (int j = 0; j < 8; ++j) m += hs[b][(size_t)i * 8 + j];
    m *= 0.125f;
    s[b] = 1.0f / (1.0f + expf(-m));
  }
  float mx = fmaxf(s[0], fmaxf(s[1], s[2]));
  float e0 = expf(s[0] - mx), e1 = expf(s[1] - mx), e2 = expf(s[2] - mx);
  float inv = 1.0f / (e0 + e1 + e2);
  g[i * 3 + 0] = e0 * inv; g[i * 3 + 1] = e1 * inv; g[i * 3 + 2] = e2 * inv;
}

__global__ __launch_bounds__(256)
void fuse_gate_pool3_kernel(const float* __restrict__ d1, const float* __restrict__ d2,
                            const float* __restrict__ e3, const float* __restrict__ g,
                            float* __restrict__ out) {
  int gid = blockIdx.x * 256 + threadIdx.x;
  if (gid >= ND * 128) return;
  int i = gid >> 7, j = gid & 127;
  float g0 = g[i * 3 + 0], g1 = g[i * 3 + 1], g2 = g[i * 3 + 2];
  float s = 0.f;
#pragma unroll
  for (int t = 0; t < 3; ++t) {
    int idx = 3 * j + t;
    float v;
    if (idx < 128)      v = d1[(size_t)i * 128 + idx] * g0;
    else if (idx < 256) v = d2[(size_t)i * 128 + idx - 128] * g1;
    else                v = e3[(size_t)i * 128 + idx - 256] * g2;
    s += v;
  }
  out[gid] = s * (1.0f / 3.0f);
}

__global__ __launch_bounds__(128)
void colstats_kernel(const float* __restrict__ X, float* __restrict__ sum,
                     float* __restrict__ sumsq, int N) {
  int j = threadIdx.x;
  int r0 = blockIdx.x * 64;
  int r1 = min(r0 + 64, N);
  float s = 0.f, q = 0.f;
  for (int r = r0; r < r1; ++r) {
    float v = X[(size_t)r * 128 + j];
    s += v; q += v * v;
  }
  atomicAdd(&sum[j], s);
  atomicAdd(&sumsq[j], q);
}

__global__ __launch_bounds__(128)
void bn_finalize_kernel(const float* __restrict__ sum, const float* __restrict__ sumsq,
                        float* __restrict__ mu, float* __restrict__ rsig, float invN) {
  int j = threadIdx.x;
  float m = sum[j] * invN;
  float var = sumsq[j] * invN - m * m;
  mu[j] = m;
  rsig[j] = rsqrtf(var + 1e-5f);
}

__global__ __launch_bounds__(256)
void bn_apply_kernel(const float* __restrict__ X, const float* __restrict__ mu,
                     const float* __restrict__ rsig, float* __restrict__ Y, int total) {
  int gid = blockIdx.x * 256 + threadIdx.x;
  if (gid >= total) return;
  int j = gid & 127;
  Y[gid] = (X[gid] - mu[j]) * rsig[j];
}

__global__ __launch_bounds__(256)
void combine_repeat_kernel(const float* __restrict__ convraw, const float* __restrict__ mu,
                           const float* __restrict__ rsig, const float* __restrict__ xd_bn,
                           float* __restrict__ xd256) {
  int gid = blockIdx.x * 256 + threadIdx.x;
  if (gid >= ND * 128) return;
  int i = gid >> 7, j = gid & 127;
  float v = (convraw[gid] - mu[j]) * rsig[j] + xd_bn[gid];
  xd256[(size_t)i * 256 + 2 * j]     = v;
  xd256[(size_t)i * 256 + 2 * j + 1] = v;
}

__global__ __launch_bounds__(256)
void xp_pool_kernel(const float* __restrict__ xp, float* __restrict__ out) {
  int gid = blockIdx.x * 256 + threadIdx.x;
  if (gid >= NP * 256) return;
  int i = gid >> 8, j = gid & 255;
  const float* p = xp + (size_t)i * 512 + 2 * j;
  out[gid] = 0.5f * (p[0] + p[1]);
}

__global__ __launch_bounds__(256)
void edge_count_kernel(const int* __restrict__ dst, float* __restrict__ cnt) {
  int e = blockIdx.x * 256 + threadIdx.x;
  if (e >= NE) return;
  atomicAdd(&cnt[dst[e]], 1.0f);
}

__global__ __launch_bounds__(256)
void rcp_kernel(const float* __restrict__ cnt, float* __restrict__ rcp, int n) {
  int i = blockIdx.x * 256 + threadIdx.x;
  if (i >= n) return;
  rcp[i] = 1.0f / fmaxf(cnt[i], 1.0f);
}

__global__ __launch_bounds__(256)
void scatter_add_kernel(const float* __restrict__ src, const int* __restrict__ sidx,
                        const int* __restrict__ didx, float* __restrict__ acc, int K) {
  int kq = K >> 2;
  long gid = (long)blockIdx.x * 256 + threadIdx.x;
  if (gid >= (long)NE * kq) return;
  int e = (int)(gid / kq);
  int c = (int)(gid - (long)e * kq) * 4;
  int s = sidx[e], d = didx[e];
  const float4 v = *reinterpret_cast<const float4*>(src + (size_t)s * K + c);
  float* dp = acc + (size_t)d * K + c;
  atomicAdd(dp + 0, v.x); atomicAdd(dp + 1, v.y);
  atomicAdd(dp + 2, v.z); atomicAdd(dp + 3, v.w);
}

__global__ __launch_bounds__(256)
void scale_rows_kernel(float* __restrict__ X, const float* __restrict__ rcp,
                       int total, int K) {
  int gid = blockIdx.x * 256 + threadIdx.x;
  if (gid >= total) return;
  X[gid] *= rcp[gid / K];
}

// ---------------- fused per-row conv net (1->32->64->128, 2-tap, maxpool4) ----
__global__ __launch_bounds__(128)
void conv_net_kernel(const float* __restrict__ xin,
                     const float* __restrict__ w1, const float* __restrict__ b1,
                     const float* __restrict__ w2, const float* __restrict__ b2,
                     const float* __restrict__ w3, const float* __restrict__ b3,
                     float* __restrict__ convraw, int N) {
  __shared__ float xs[128];
  __shared__ float h1[31 * 32];
  __shared__ float h2[7 * 64];
  __shared__ float w2s[32 * 64 * 2];
  for (int idx = threadIdx.x; idx < 32 * 64 * 2; idx += 128) w2s[idx] = w2[idx];
  __syncthreads();

  for (int row = blockIdx.x; row < N; row += gridDim.x) {
    xs[threadIdx.x] = xin[(size_t)row * 128 + threadIdx.x];
    __syncthreads();

    for (int idx = threadIdx.x; idx < 992; idx += 128) {
      int c = idx & 31, t = idx >> 5;
      float wa = w1[c * 2 + 0], wb = w1[c * 2 + 1], bb = b1[c];
      float m = 0.f;
#pragma unroll
      for (int p = 0; p < 4; ++p) {
        int pos = 4 * t + p;
        float y = wb * xs[pos] + wa * xs[pos + 1] + bb;
        m = fmaxf(m, fmaxf(y, 0.f));
      }
      h1[t * 32 + c] = m;
    }
    __syncthreads();

    for (int idx = threadIdx.x; idx < 448; idx += 128) {
      int c = idx & 63, tp = idx >> 6;
      float bb = b2[c];
      float m = 0.f;
      for (int t = 4 * tp; t < 4 * tp + 4; ++t) {
        float acc = bb;
        for (int i = 0; i < 32; ++i) {
          acc += w2s[(i * 64 + c) * 2 + 1] * h1[t * 32 + i]
               + w2s[(i * 64 + c) * 2 + 0] * h1[(t + 1) * 32 + i];
        }
        m = fmaxf(m, fmaxf(acc, 0.f));
      }
      h2[tp * 64 + c] = m;
    }
    __syncthreads();

    for (int c = threadIdx.x; c < 128; c += 128) {
      float bb = b3[c];
      float m = 0.f;
#pragma unroll
      for (int t = 0; t < 4; ++t) {
        float acc = bb;
        for (int i = 0; i < 64; ++i) {
          acc += w3[(i * 128 + c) * 2 + 1] * h2[t * 64 + i]
               + w3[(i * 128 + c) * 2 + 0] * h2[(t + 1) * 64 + i];
        }
        m = fmaxf(m, fmaxf(acc, 0.f));
      }
      convraw[(size_t)row * 128 + c] = m;
    }
    __syncthreads();
  }
}

// ---------------------------------------------------------------------------
extern "C" void kernel_launch(void* const* d_in, const int* in_sizes, int n_in,
                              void* d_out, int out_size, void* d_ws, size_t ws_size,
                              hipStream_t stream) {
  (void)in_sizes; (void)n_in; (void)out_size; (void)ws_size;

  const float* x_drug = (const float*)d_in[0];
  const float* x_prot = (const float*)d_in[1];
  const float* x2d    = (const float*)d_in[2];
  const float* x3d    = (const float*)d_in[3];
  const int*   e_pd   = (const int*)d_in[4];
  const int*   e_dp   = (const int*)d_in[5];
  const float* gW1 = (const float*)d_in[6];  const float* gb1 = (const float*)d_in[7];
  const float* gW2 = (const float*)d_in[8];  const float* gb2 = (const float*)d_in[9];
  const float* gW3 = (const float*)d_in[10]; const float* gb3 = (const float*)d_in[11];
  const float* cw1 = (const float*)d_in[12]; const float* cb1 = (const float*)d_in[13];
  const float* cw2 = (const float*)d_in[14]; const float* cb2 = (const float*)d_in[15];
  const float* cw3 = (const float*)d_in[16]; const float* cb3 = (const float*)d_in[17];

  char* ws = (char*)d_ws;
  size_t off = 0;
  auto alloc = [&](size_t nfloats) -> float* {
    float* p = (float*)(ws + off);
    off += ((nfloats * 4 + 255) / 256) * 256;
    return p;
  };
  float* d1f      = alloc((size_t)ND * 128);
  float* d2f      = alloc((size_t)ND * 128);   // d1f..d2f dead by SAGE time
  float* mean_d   = d1f;                       // alias: ND x 256
  float* h1f      = alloc((size_t)ND * 64);
  float* h2f      = alloc((size_t)ND * 32);
  float* h3f      = alloc((size_t)ND * 8 * 3);
  float* gg       = alloc((size_t)ND * 3);
  float* xd128raw = alloc((size_t)ND * 128);
  float* xd_bn    = alloc((size_t)ND * 128);   // raw+bn dead by SAGE time
  float* mean_p   = xd128raw;                  // alias: NP x 256 fits in ND x 256
  float* convraw  = alloc((size_t)ND * 128);
  float* stats    = alloc(8 * 128);
  float* xd256    = alloc((size_t)ND * 256);
  float* xp256    = alloc((size_t)NP * 256);
  float* cnt_d    = alloc(ND);  float* rcp_d = alloc(ND);
  float* cnt_p    = alloc(NP);  float* rcp_p = alloc(NP);
  float* xdA      = alloc((size_t)ND * 128);
  float* xpA      = alloc((size_t)NP * 128);
  float* xdB      = alloc((size_t)ND * 64);
  float* xpB      = alloc((size_t)NP * 64);

  auto gemm = [&](const float* A0, const float* W0, int K0,
                  const float* A1, const float* W1, int K1,
                  const float* bias, float* out, int N, int M, int act) {
    dim3 blk(32, 4);
    dim3 grd((N + 63) / 64, 1);
    int ct = (M + 15) / 16;
    switch (ct) {
      case 1: wmma_gemm_kernel<1><<<grd, blk, 0, stream>>>(A0, W0, K0, A1, W1, K1, bias, out, N, M, act); break;
      case 2: wmma_gemm_kernel<2><<<grd, blk, 0, stream>>>(A0, W0, K0, A1, W1, K1, bias, out, N, M, act); break;
      case 4: wmma_gemm_kernel<4><<<grd, blk, 0, stream>>>(A0, W0, K0, A1, W1, K1, bias, out, N, M, act); break;
      default: wmma_gemm_kernel<8><<<grd, blk, 0, stream>>>(A0, W0, K0, A1, W1, K1, bias, out, N, M, act); break;
    }
  };

  // ---- drug pooling + gate MLP (WMMA) ----
  pool4_kernel<<<(ND * 128 + 255) / 256, 256, 0, stream>>>(x_drug, x2d, d1f, d2f);
  const float* branch_in[3] = { d1f, d2f, x3d };
  for (int br = 0; br < 3; ++br) {
    gemm(branch_in[br], gW1, 128, nullptr, nullptr, 0, gb1, h1f, ND, 64, 1);
    gemm(h1f,           gW2,  64, nullptr, nullptr, 0, gb2, h2f, ND, 32, 1);
    gemm(h2f,           gW3,  32, nullptr, nullptr, 0, gb3,
         h3f + (size_t)br * ND * 8, ND, 8, 1);
  }
  gate_softmax_kernel<<<(ND + 255) / 256, 256, 0, stream>>>(
      h3f, h3f + (size_t)ND * 8, h3f + (size_t)ND * 16, gg);
  fuse_gate_pool3_kernel<<<(ND * 128 + 255) / 256, 256, 0, stream>>>(
      d1f, d2f, x3d, gg, xd128raw);

  // ---- batchnorm 1 ----
  hipMemsetAsync(stats, 0, 8 * 128 * sizeof(float), stream);
  colstats_kernel<<<(ND + 63) / 64, 128, 0, stream>>>(xd128raw, stats, stats + 128, ND);
  bn_finalize_kernel<<<1, 128, 0, stream>>>(stats, stats + 128, stats + 256, stats + 384,
                                            1.0f / (float)ND);
  bn_apply_kernel<<<(ND * 128 + 255) / 256, 256, 0, stream>>>(
      xd128raw, stats + 256, stats + 384, xd_bn, ND * 128);

  // ---- conv net + batchnorm 2 + residual + repeat x2 ----
  conv_net_kernel<<<2048, 128, 0, stream>>>(xd_bn, cw1, cb1, cw2, cb2, cw3, cb3,
                                            convraw, ND);
  colstats_kernel<<<(ND + 63) / 64, 128, 0, stream>>>(convraw, stats + 512, stats + 640, ND);
  bn_finalize_kernel<<<1, 128, 0, stream>>>(stats + 512, stats + 640, stats + 768,
                                            stats + 896, 1.0f / (float)ND);
  combine_repeat_kernel<<<(ND * 128 + 255) / 256, 256, 0, stream>>>(
      convraw, stats + 768, stats + 896, xd_bn, xd256);
  xp_pool_kernel<<<(NP * 256 + 255) / 256, 256, 0, stream>>>(x_prot, xp256);

  // ---- edge degree (constant across layers) ----
  hipMemsetAsync(cnt_d, 0, ND * sizeof(float), stream);
  hipMemsetAsync(cnt_p, 0, NP * sizeof(float), stream);
  edge_count_kernel<<<(NE + 255) / 256, 256, 0, stream>>>(e_pd + NE, cnt_d);
  edge_count_kernel<<<(NE + 255) / 256, 256, 0, stream>>>(e_dp + NE, cnt_p);
  rcp_kernel<<<(ND + 255) / 256, 256, 0, stream>>>(cnt_d, rcp_d, ND);
  rcp_kernel<<<(NP + 255) / 256, 256, 0, stream>>>(cnt_p, rcp_p, NP);

  // ---- 3 SAGE layers: scatter-mean + fused two-source WMMA GEMM + leaky ----
  float* xd_out = (float*)d_out;                        // 30000 x 32
  float* xp_out = xd_out + (size_t)ND * 32;             // 20000 x 32
  const int Ks[3] = { 256, 128, 64 };
  const int Ms[3] = { 128, 64, 32 };
  float* xdN[3] = { xdA, xdB, xd_out };
  float* xpN[3] = { xpA, xpB, xp_out };
  const float* xd_cur = xd256;
  const float* xp_cur = xp256;

  for (int l = 0; l < 3; ++l) {
    int K = Ks[l], M = Ms[l];
    const float* pdWl = (const float*)d_in[18 + l * 6 + 0];
    const float* pdbl = (const float*)d_in[18 + l * 6 + 1];
    const float* pdWr = (const float*)d_in[18 + l * 6 + 2];
    const float* dpWl = (const float*)d_in[18 + l * 6 + 3];
    const float* dpbl = (const float*)d_in[18 + l * 6 + 4];
    const float* dpWr = (const float*)d_in[18 + l * 6 + 5];

    hipMemsetAsync(mean_d, 0, (size_t)ND * K * sizeof(float), stream);
    hipMemsetAsync(mean_p, 0, (size_t)NP * K * sizeof(float), stream);

    long nthr = (long)NE * (K >> 2);
    scatter_add_kernel<<<(unsigned)((nthr + 255) / 256), 256, 0, stream>>>(
        xp_cur, e_pd, e_pd + NE, mean_d, K);
    scatter_add_kernel<<<(unsigned)((nthr + 255) / 256), 256, 0, stream>>>(
        xd_cur, e_dp, e_dp + NE, mean_p, K);
    scale_rows_kernel<<<(ND * K + 255) / 256, 256, 0, stream>>>(mean_d, rcp_d, ND * K, K);
    scale_rows_kernel<<<(NP * K + 255) / 256, 256, 0, stream>>>(mean_p, rcp_p, NP * K, K);

    gemm(mean_d, pdWl, K, xd_cur, pdWr, K, pdbl, xdN[l], ND, M, 2);
    gemm(mean_p, dpWl, K, xp_cur, dpWr, K, dpbl, xpN[l], NP, M, 2);

    xd_cur = xdN[l];
    xp_cur = xpN[l];
  }
}
